// MultiHeadAttention_17678085390427
// MI455X (gfx1250) — matrix operand order repaired and checked by reading, not compiled, not run on hardware
//
#include <hip/hip_runtime.h>

// ---------------------------------------------------------------------------
// Types
// ---------------------------------------------------------------------------
typedef __bf16 bf16;
typedef __attribute__((ext_vector_type(16))) __bf16 v16bf;
typedef __attribute__((ext_vector_type(8)))  __bf16 bf16x8;
typedef __attribute__((ext_vector_type(4)))  __bf16 bf16x4;
typedef __attribute__((ext_vector_type(8)))  float   v8f;
typedef __attribute__((ext_vector_type(4)))  float   f32x4;

#define BATCH  4
#define SEQ    2048
#define DMODEL 1024
#define NHEAD  16
#define HD     64

#define GM (BATCH * SEQ)   // 8192 rows for all projections
#define GN DMODEL          // 1024
#define GK DMODEL          // 1024

__device__ __forceinline__ v8f wmma_bf16(v16bf a, v16bf b, v8f c) {
  return __builtin_amdgcn_wmma_f32_16x16x32_bf16(false, a, false, b,
                                                 (short)0, c, false, false);
}

__device__ __forceinline__ v16bf pack16(bf16x8 lo, bf16x8 hi) {
  v16bf r;
#pragma unroll
  for (int i = 0; i < 8; ++i) { r[i] = lo[i]; r[i + 8] = hi[i]; }
  return r;
}

// Async global -> LDS copy, 16B per lane (ASYNCcnt-tracked, no VGPR staging).
// Flat shared pointers truncate to the 32-bit LDS offset (ISA 10.2 aperture).
__device__ __forceinline__ void async_b128(const bf16* g, bf16* l) {
  unsigned lds = (unsigned)(uintptr_t)l;
  asm volatile("global_load_async_to_lds_b128 %0, %1, off"
               :: "v"(lds), "v"(g)
               : "memory");
}
__device__ __forceinline__ void wait_async() {
  asm volatile("s_wait_asynccnt 0" ::: "memory");
}

// ---------------------------------------------------------------------------
// fp32 -> bf16 conversion (vectorized)
// ---------------------------------------------------------------------------
__global__ __launch_bounds__(256) void f32_to_bf16_kernel(const float* __restrict__ src,
                                                          bf16* __restrict__ dst, int n) {
  int i = (blockIdx.x * blockDim.x + threadIdx.x) * 4;
  if (i < n) {
    f32x4 f = *(const f32x4*)(src + i);
    bf16x4 o;
#pragma unroll
    for (int j = 0; j < 4; ++j) o[j] = (bf16)f[j];
    *(bf16x4*)(dst + i) = o;
  }
}

// ---------------------------------------------------------------------------
// Tiled GEMM + bias:  C[M,N] = A[M,K](bf16) * B[K,N](bf16) + bias[N](f32)
// Block tile 128x128x32, 256 threads = 8 waves (2x4), wave tile 64x32.
// Double-buffered LDS staged via global_load_async_to_lds_b128.
// ---------------------------------------------------------------------------
template <bool OUT_BF16>
__global__ __launch_bounds__(256) void gemm_bias_kernel(const bf16* __restrict__ A,
                                                        const bf16* __restrict__ B,
                                                        const float* __restrict__ bias,
                                                        bf16* __restrict__ Cb,
                                                        float* __restrict__ Cf) {
  __shared__ alignas(16) bf16 As[2][128][40];   // 32 + 8 pad (80B row, 16B aligned)
  __shared__ alignas(16) bf16 Bs[2][32][136];   // 128 + 8 pad (272B row, 16B aligned)

  const int tid  = threadIdx.x;
  const int wave = tid >> 5;
  const int lane = tid & 31;
  const int wm   = wave >> 2;        // 0..1  (64 rows each)
  const int wn   = wave & 3;         // 0..3  (32 cols each)
  const int bm   = blockIdx.y * 128;
  const int bn   = blockIdx.x * 128;

  const int mlane = lane & 15;
  const int kbase = (lane < 16) ? 0 : 8;
  const int half  = lane >> 4;

  v8f acc[4][2];
#pragma unroll
  for (int i = 0; i < 4; ++i)
#pragma unroll
    for (int j = 0; j < 2; ++j) acc[i][j] = (v8f)(0.0f);

  auto stage = [&](int buf, int k0) {
    // A tile: 128x32, one async 16B chunk per thread x2
#pragma unroll
    for (int i = 0; i < 2; ++i) {
      int chunk = tid + 256 * i;           // 0..511
      int row   = chunk >> 2;              // 0..127
      int col   = (chunk & 3) * 8;         // 0,8,16,24
      async_b128(A + (size_t)(bm + row) * GK + k0 + col, &As[buf][row][col]);
    }
    // B tile: 32x128
#pragma unroll
    for (int i = 0; i < 2; ++i) {
      int chunk = tid + 256 * i;           // 0..511
      int row   = chunk >> 4;              // 0..31
      int col   = (chunk & 15) * 8;        // 0..120
      async_b128(B + (size_t)(k0 + row) * GN + bn + col, &Bs[buf][row][col]);
    }
  };

  stage(0, 0);
  wait_async();
  __syncthreads();

  for (int k0 = 0; k0 < GK; k0 += 32) {
    const int cur = (k0 >> 5) & 1;
    if (k0 + 32 < GK) stage(cur ^ 1, k0 + 32);   // prefetch next tile (async)

    // ---- A fragments (4 M-tiles x 16x32) ----
    v16bf af[4];
#pragma unroll
    for (int mt = 0; mt < 4; ++mt) {
      const bf16* p = &As[cur][wm * 64 + mt * 16 + mlane][0];
      af[mt] = pack16(*(const bf16x8*)(p + kbase),
                      *(const bf16x8*)(p + 16 + kbase));
    }
    // ---- B fragments (2 N-tiles x 32x16): lane%16 = n col, elements over k ----
    v16bf bfr[2];
#pragma unroll
    for (int nt = 0; nt < 2; ++nt) {
      const bf16* p = &Bs[cur][0][wn * 32 + nt * 16 + mlane];
#pragma unroll
      for (int i = 0; i < 8; ++i) {
        bfr[nt][i]     = p[(kbase + i) * 136];
        bfr[nt][i + 8] = p[(16 + kbase + i) * 136];
      }
    }
#pragma unroll
    for (int mt = 0; mt < 4; ++mt)
#pragma unroll
      for (int nt = 0; nt < 2; ++nt)
        acc[mt][nt] = wmma_bf16(af[mt], bfr[nt], acc[mt][nt]);

    wait_async();      // our prefetch landed
    __syncthreads();   // everyone done reading cur / writing nxt
  }

  // ---- epilogue: + bias, store ----
#pragma unroll
  for (int mt = 0; mt < 4; ++mt) {
#pragma unroll
    for (int nt = 0; nt < 2; ++nt) {
      int col  = bn + wn * 32 + nt * 16 + mlane;
      float bv = bias[col];
#pragma unroll
      for (int r = 0; r < 8; ++r) {
        int row = bm + wm * 64 + mt * 16 + r + 8 * half;
        float v = acc[mt][nt][r] + bv;
        if (OUT_BF16)
          Cb[(size_t)row * GN + col] = (bf16)v;
        else
          Cf[(size_t)row * GN + col] = v;
      }
    }
  }
}

// ---------------------------------------------------------------------------
// Causal flash attention, one head-slice per blockIdx.y, 64 query rows per
// blockIdx.x. 128 threads = 4 waves, each wave owns 16 query rows.
// Q/K/V/O are contiguous [SEQ][HD] bf16 matrices per (b,h) (plain reshape).
// ---------------------------------------------------------------------------
__global__ __launch_bounds__(128) void attn_kernel(const bf16* __restrict__ Q,
                                                   const bf16* __restrict__ K,
                                                   const bf16* __restrict__ V,
                                                   bf16* __restrict__ O) {
  __shared__ alignas(16) bf16 Kt[64][40];      // K^T: [d][key], 32 keys + pad
  __shared__ alignas(16) bf16 Vt[32][72];      // V:   [key][d], 64 d + pad
  __shared__ alignas(16) bf16 Pl[4][16][32];   // per-wave P scratch

  const int qb  = blockIdx.x;           // 0..31
  const int bh  = blockIdx.y;           // 0..63
  const size_t base = (size_t)bh * SEQ * HD;
  const bf16* Qp = Q + base;
  const bf16* Kp = K + base;
  const bf16* Vp = V + base;
  bf16*       Op = O + base;

  const int tid  = threadIdx.x;
  const int wave = tid >> 5;
  const int lane = tid & 31;
  const int mlane = lane & 15;
  const int kbase = (lane < 16) ? 0 : 8;
  const int half  = lane >> 4;
  const int q0    = qb * 64 + wave * 16;

  // Q fragments: 16 rows x 64 d  ->  two 16x32 A fragments
  v16bf aq[2];
#pragma unroll
  for (int f = 0; f < 2; ++f) {
    const bf16* p = Qp + (size_t)(q0 + mlane) * HD + f * 32;
    aq[f] = pack16(*(const bf16x8*)(p + kbase),
                   *(const bf16x8*)(p + 16 + kbase));
  }

  float rm[8], rl[8];
#pragma unroll
  for (int r = 0; r < 8; ++r) { rm[r] = -1e30f; rl[r] = 0.0f; }
  v8f o[4];
#pragma unroll
  for (int j = 0; j < 4; ++j) o[j] = (v8f)(0.0f);

  const int ktiles = qb * 2 + 2;        // keys 0 .. qb*64+63 in steps of 32
  for (int t = 0; t < ktiles; ++t) {
    const int kt = t * 32;
    __syncthreads();                    // previous tile fully consumed
    // ---- cooperative load: V async to LDS, K transposed via VGPRs ----
#pragma unroll
    for (int i = 0; i < 2; ++i) {
      int chunk = tid + 128 * i;        // 0..255
      int krow  = chunk >> 3;           // 0..31
      int d0    = (chunk & 7) * 8;      // 0..56
      async_b128(Vp + (size_t)(kt + krow) * HD + d0, &Vt[krow][d0]);
      bf16x8 kv = *(const bf16x8*)(Kp + (size_t)(kt + krow) * HD + d0);
#pragma unroll
      for (int j = 0; j < 8; ++j) Kt[d0 + j][krow] = kv[j];
    }
    wait_async();
    __syncthreads();

    // ---- scores: S[16 x 32] = Q(16x64) * K^T(64x32), two chained WMMAs ----
    v8f sc[2];
#pragma unroll
    for (int nt = 0; nt < 2; ++nt) {
      sc[nt] = (v8f)(0.0f);
#pragma unroll
      for (int ch = 0; ch < 2; ++ch) {
        const bf16* p = &Kt[ch * 32][nt * 16 + mlane];
        v16bf bk;
#pragma unroll
        for (int i = 0; i < 8; ++i) {
          bk[i]     = p[(kbase + i) * 40];
          bk[i + 8] = p[(16 + kbase + i) * 40];
        }
        sc[nt] = wmma_bf16(aq[ch], bk, sc[nt]);
      }
    }

    // ---- scale + causal mask ----
#pragma unroll
    for (int nt = 0; nt < 2; ++nt) {
      int kcol = kt + nt * 16 + mlane;
#pragma unroll
      for (int r = 0; r < 8; ++r) {
        int qrow = q0 + r + 8 * half;
        float s  = sc[nt][r] * 0.125f;  // 1/sqrt(64)
        sc[nt][r] = (kcol <= qrow) ? s : -1e30f;
      }
    }

    // ---- online softmax (row reductions over 16-lane halves) ----
    float fac[8];
#pragma unroll
    for (int r = 0; r < 8; ++r) {
      float v = fmaxf(sc[0][r], sc[1][r]);
#pragma unroll
      for (int m = 1; m <= 8; m <<= 1) v = fmaxf(v, __shfl_xor(v, m, 16));
      float mn = fmaxf(rm[r], v);
      fac[r]   = __expf(rm[r] - mn);
      rm[r]    = mn;
      float p0 = __expf(sc[0][r] - mn);
      float p1 = __expf(sc[1][r] - mn);
      sc[0][r] = p0;
      sc[1][r] = p1;
      float ps = p0 + p1;
#pragma unroll
      for (int m = 1; m <= 8; m <<= 1) ps += __shfl_xor(ps, m, 16);
      rl[r] = rl[r] * fac[r] + ps;
#pragma unroll
      for (int j = 0; j < 4; ++j) o[j][r] *= fac[r];
    }

    // ---- P -> LDS (C layout) -> A fragment layout ----
#pragma unroll
    for (int nt = 0; nt < 2; ++nt)
#pragma unroll
      for (int r = 0; r < 8; ++r)
        Pl[wave][r + 8 * half][nt * 16 + mlane] = (bf16)sc[nt][r];

    asm volatile("s_wait_dscnt 0" ::: "memory");  // per-wave LDS RAW fence

    const bf16* pp = &Pl[wave][mlane][0];
    v16bf ap = pack16(*(const bf16x8*)(pp + kbase),
                      *(const bf16x8*)(pp + 16 + kbase));

    // ---- O += P(16x32) * V(32x64) ----
#pragma unroll
    for (int j = 0; j < 4; ++j) {
      const bf16* p = &Vt[0][j * 16 + mlane];
      v16bf bv;
#pragma unroll
      for (int i = 0; i < 8; ++i) {
        bv[i]     = p[(kbase + i) * 72];
        bv[i + 8] = p[(16 + kbase + i) * 72];
      }
      o[j] = wmma_bf16(ap, bv, o[j]);
    }
  }

  // ---- normalize + store ----
#pragma unroll
  for (int r = 0; r < 8; ++r) {
    float inv = (rl[r] > 0.0f) ? (1.0f / rl[r]) : 0.0f;
    int qrow  = q0 + r + 8 * half;
#pragma unroll
    for (int j = 0; j < 4; ++j)
      Op[(size_t)qrow * HD + j * 16 + mlane] = (bf16)(o[j][r] * inv);
  }
}

// ---------------------------------------------------------------------------
// Host-side launch
// ---------------------------------------------------------------------------
extern "C" void kernel_launch(void* const* d_in, const int* in_sizes, int n_in,
                              void* d_out, int out_size, void* d_ws, size_t ws_size,
                              hipStream_t stream) {
  (void)in_sizes; (void)n_in; (void)out_size; (void)ws_size;

  const float* x  = (const float*)d_in[0];
  const float* Wq = (const float*)d_in[1];
  const float* bq = (const float*)d_in[2];
  const float* Wk = (const float*)d_in[3];
  const float* bk = (const float*)d_in[4];
  const float* Wv = (const float*)d_in[5];
  const float* bv = (const float*)d_in[6];
  const float* Wo = (const float*)d_in[7];
  const float* bo = (const float*)d_in[8];
  float* out = (float*)d_out;

  const size_t XN = (size_t)GM * GK;       // 8192*1024
  const size_t WN = (size_t)GK * GN;       // 1024*1024

  bf16* ws  = (bf16*)d_ws;
  bf16* Xb  = ws;              ws += XN;
  bf16* Wqb = ws;              ws += WN;
  bf16* Wkb = ws;              ws += WN;
  bf16* Wvb = ws;              ws += WN;
  bf16* Wob = ws;              ws += WN;
  bf16* Qb  = ws;              ws += XN;
  bf16* Kb  = ws;              ws += XN;
  bf16* Vb  = ws;              ws += XN;
  bf16* Ab  = ws;              ws += XN;

  // 1) fp32 -> bf16 conversions
  {
    int n = (int)XN;
    f32_to_bf16_kernel<<<(n / 4 + 255) / 256, 256, 0, stream>>>(x, Xb, n);
    int nw = (int)WN;
    dim3 g((nw / 4 + 255) / 256);
    f32_to_bf16_kernel<<<g, 256, 0, stream>>>(Wq, Wqb, nw);
    f32_to_bf16_kernel<<<g, 256, 0, stream>>>(Wk, Wkb, nw);
    f32_to_bf16_kernel<<<g, 256, 0, stream>>>(Wv, Wvb, nw);
    f32_to_bf16_kernel<<<g, 256, 0, stream>>>(Wo, Wob, nw);
  }

  // 2) Q/K/V projections (bf16 out)
  {
    dim3 grid(GN / 128, GM / 128);   // (8, 64)
    gemm_bias_kernel<true><<<grid, 256, 0, stream>>>(Xb, Wqb, bq, Qb, nullptr);
    gemm_bias_kernel<true><<<grid, 256, 0, stream>>>(Xb, Wkb, bk, Kb, nullptr);
    gemm_bias_kernel<true><<<grid, 256, 0, stream>>>(Xb, Wvb, bv, Vb, nullptr);
  }

  // 3) causal attention per contiguous head slice
  {
    dim3 grid(SEQ / 64, BATCH * NHEAD);  // (32, 64)
    attn_kernel<<<grid, 128, 0, stream>>>(Qb, Kb, Vb, Ab);
  }

  // 4) output projection (fp32 out)
  {
    dim3 grid(GN / 128, GM / 128);
    gemm_bias_kernel<false><<<grid, 256, 0, stream>>>(Ab, Wob, bo, nullptr, out);
  }
}
